// CausalSelfAttention_7679401526026
// MI455X (gfx1250) — compile-verified, matrix operand order checked by reading, not run on hardware
//
#include <hip/hip_runtime.h>
#include <hip/hip_bf16.h>

typedef __attribute__((ext_vector_type(16))) _Float16 v16h;
typedef __attribute__((ext_vector_type(8)))  _Float16 v8h;
typedef __attribute__((ext_vector_type(8)))  float    v8f;

#define DEV __device__ __forceinline__

static constexpr int Bb = 4, T = 2048, C = 1024, H = 16, HD = 64;
static constexpr int M     = Bb * T;    // 8192 token rows
static constexpr int N_QKV = 3 * C;     // 3072
static constexpr float SCALE = -1.0f / 16.0f;  // -1/(2*sqrt(hd)), hd=64

// Load a 16x32 f16 A/B fragment for one lane from a row-major buffer.
// Per CDNA5 ISA 7.12.2 (16-bit A 16x32): lane L (half h=L>>4) holds
// elements e=0..7 -> k = kb + 8h + e, e=8..15 -> k = kb + 16 + 8h + (e-8).
// Caller pre-offsets `rowbase` to the lane's row (A: m = lane&15) or
// column (B: n = lane&15, from a transposed [N][K] buffer).
DEV v16h load_frag(const _Float16* rowbase, int kb, int lane) {
  const int h = (lane >> 4) & 1;
  v8h lo = *(const v8h*)(rowbase + kb + 8 * h);
  v8h hi = *(const v8h*)(rowbase + kb + 16 + 8 * h);
  v16h r;
#pragma unroll
  for (int i = 0; i < 8; ++i) { r[i] = lo[i]; r[i + 8] = hi[i]; }
  return r;
}

DEV v8f wmma_f16(v16h a, v16h b, v8f c) {
  return __builtin_amdgcn_wmma_f32_16x16x32_f16(false, a, false, b, (short)0, c,
                                                false, false);
}

// Software-pipelined 32x64 wave-tile GEMM inner loop over K=C.
// acc[mt*4+j] accumulates (16 rows of a0r/a1r) x (16 cols of br_j).
DEV void gemm_loop_32x64(const _Float16* a0r, const _Float16* a1r,
                         const _Float16* br0, const _Float16* br1,
                         const _Float16* br2, const _Float16* br3,
                         v8f acc[8], int lane) {
  v16h a0 = load_frag(a0r, 0, lane), a1 = load_frag(a1r, 0, lane);
  v16h b0 = load_frag(br0, 0, lane), b1 = load_frag(br1, 0, lane);
  v16h b2 = load_frag(br2, 0, lane), b3 = load_frag(br3, 0, lane);
#pragma unroll 2
  for (int kb = 0; kb < C; kb += 32) {
    const int kn = (kb + 32 < C) ? kb + 32 : 0;  // last iter: harmless reload
    // prefetch next step's fragments first -> loads overlap current WMMAs
    v16h na0 = load_frag(a0r, kn, lane), na1 = load_frag(a1r, kn, lane);
    v16h nb0 = load_frag(br0, kn, lane), nb1 = load_frag(br1, kn, lane);
    v16h nb2 = load_frag(br2, kn, lane), nb3 = load_frag(br3, kn, lane);
    acc[0] = wmma_f16(a0, b0, acc[0]);
    acc[1] = wmma_f16(a0, b1, acc[1]);
    acc[2] = wmma_f16(a0, b2, acc[2]);
    acc[3] = wmma_f16(a0, b3, acc[3]);
    acc[4] = wmma_f16(a1, b0, acc[4]);
    acc[5] = wmma_f16(a1, b1, acc[5]);
    acc[6] = wmma_f16(a1, b2, acc[6]);
    acc[7] = wmma_f16(a1, b3, acc[7]);
    a0 = na0; a1 = na1; b0 = nb0; b1 = nb1; b2 = nb2; b3 = nb3;
  }
}

// ---------------- prep kernels ----------------

__global__ void cvt_f32_f16(const float* __restrict__ in,
                            _Float16* __restrict__ outh, int n) {
  for (int i = blockIdx.x * blockDim.x + threadIdx.x; i < n;
       i += gridDim.x * blockDim.x)
    outh[i] = (_Float16)in[i];
}

// in: f32 [R][Cc] row-major  ->  out: f16 [Cc][R]
__global__ __launch_bounds__(1024)
void transpose_f32_f16(const float* __restrict__ in,
                       _Float16* __restrict__ outh, int R, int Cc) {
  __shared__ float tile[32][33];
  int c = blockIdx.x * 32 + threadIdx.x;
  int r = blockIdx.y * 32 + threadIdx.y;
  tile[threadIdx.y][threadIdx.x] = in[(size_t)r * Cc + c];
  __syncthreads();
  int oc   = blockIdx.y * 32 + threadIdx.x;  // along R
  int orow = blockIdx.x * 32 + threadIdx.y;  // along Cc
  outh[(size_t)orow * R + oc] = (_Float16)tile[threadIdx.x][threadIdx.y];
}

// sum of squares over hd=64 per row
__global__ void rownorm_kernel(const _Float16* __restrict__ mh,
                               float* __restrict__ outn, int rows) {
  int i = blockIdx.x * blockDim.x + threadIdx.x;
  if (i >= rows) return;
  const _Float16* p = mh + (size_t)i * HD;
  float s = 0.f;
#pragma unroll
  for (int d = 0; d < HD; ++d) { float v = (float)p[d]; s += v * v; }
  outn[i] = s;
}

// ---------------- QKV GEMM: qkv = xh @ WtA^T + bias, split into q/k/v^T ----

__global__ __launch_bounds__(256)
void qkv_gemm_kernel(const _Float16* __restrict__ xh,
                     const _Float16* __restrict__ WtA,   // [3C][C]
                     const float* __restrict__ bias,
                     _Float16* __restrict__ qh,          // [B][H][T][hd]
                     _Float16* __restrict__ kh,          // [B][H][T][hd]
                     _Float16* __restrict__ vth) {       // [B][H][hd][T]
  const int lane = threadIdx.x & 31;
  const int wave = threadIdx.x >> 5;
  const int ln   = lane & 15;
  const int m0   = blockIdx.x * 256 + wave * 32;
  const int nb   = blockIdx.y * 64;
  v8f acc[8] = {v8f{}, v8f{}, v8f{}, v8f{}, v8f{}, v8f{}, v8f{}, v8f{}};
  gemm_loop_32x64(xh + (size_t)(m0 + ln) * C,
                  xh + (size_t)(m0 + 16 + ln) * C,
                  WtA + (size_t)(nb + ln) * C,
                  WtA + (size_t)(nb + 16 + ln) * C,
                  WtA + (size_t)(nb + 32 + ln) * C,
                  WtA + (size_t)(nb + 48 + ln) * C, acc, lane);
  const int mat  = nb >> 10;            // 0=q, 1=k, 2=v (uniform per block)
  const int head = (nb & 1023) >> 6;
#pragma unroll
  for (int mt = 0; mt < 2; ++mt) {
#pragma unroll
    for (int j = 0; j < 4; ++j) {
      int col = 16 * j + ln;            // head-dim column 0..63
      float bv = bias[nb + col];
#pragma unroll
      for (int r = 0; r < 8; ++r) {
        int m  = m0 + 16 * mt + r + 8 * (lane >> 4);
        int b_ = m >> 11, t = m & (T - 1);
        _Float16 hv = (_Float16)(acc[mt * 4 + j][r] + bv);
        size_t bhh = (size_t)(b_ * H + head);
        if (mat == 0)      qh[(bhh * T + t) * HD + col] = hv;
        else if (mat == 1) kh[(bhh * T + t) * HD + col] = hv;
        else               vth[(bhh * HD + col) * T + t] = hv;
      }
    }
  }
}

// ---------------- fused causal L2-kernel attention ----------------
// y[q] = sum_{k<=q} exp(scale*(|q|^2+|k|^2-2 q.k)) * v[k]   (no softmax norm)

__global__ __launch_bounds__(256)
void attn_kernel(const _Float16* __restrict__ qh, const _Float16* __restrict__ kh,
                 const _Float16* __restrict__ vth,
                 const float* __restrict__ q2, const float* __restrict__ k2,
                 _Float16* __restrict__ yh) {            // [B][T][C]
  __shared__ _Float16 pbuf[8][32 * 32];   // per-wave P tile, [m 0..31][key 0..31]
  const int lane = threadIdx.x & 31;
  const int wave = threadIdx.x >> 5;
  const int ln   = lane & 15;
  const int h2   = lane >> 4;
  const int bh   = blockIdx.y;            // b*H + h
  const int b_   = bh >> 4, head = bh & 15;
  const int q0   = blockIdx.x * 256 + wave * 32;   // 32 query rows per wave

  // Q fragments: 2 m-tiles x 2 contraction halves, loaded once
  v16h qa[2][2];
#pragma unroll
  for (int mt = 0; mt < 2; ++mt) {
    const _Float16* qrow = qh + ((size_t)bh * T + q0 + 16 * mt + ln) * HD;
    qa[mt][0] = load_frag(qrow, 0, lane);
    qa[mt][1] = load_frag(qrow, 32, lane);
  }
  float q2v[2][8];
#pragma unroll
  for (int mt = 0; mt < 2; ++mt)
#pragma unroll
    for (int r = 0; r < 8; ++r)
      q2v[mt][r] = q2[(size_t)bh * T + q0 + 16 * mt + 8 * h2 + r];

  const float* k2row = k2 + (size_t)bh * T;
  _Float16* myp = pbuf[wave];

  v8f yacc[2][4] = {{v8f{}, v8f{}, v8f{}, v8f{}}, {v8f{}, v8f{}, v8f{}, v8f{}}};
  const int qend = q0 + 31;               // wave-uniform causal bound
  for (int kb = 0; kb <= qend; kb += 32) {
    // ---- V fragments first: independent of S, overlap the S WMMAs/exp ----
    v16h vf[4];
#pragma unroll
    for (int j = 0; j < 4; ++j)
      vf[j] = load_frag(vth + ((size_t)bh * HD + 16 * j + ln) * T, kb, lane);
    // ---- S = Q @ K^T over 32 keys (2 key halves x 2 hd halves x 2 m) ----
    v8f sacc[2][2] = {{v8f{}, v8f{}}, {v8f{}, v8f{}}};
#pragma unroll
    for (int nh = 0; nh < 2; ++nh) {
      const _Float16* krow = kh + ((size_t)bh * T + kb + 16 * nh + ln) * HD;
#pragma unroll
      for (int kk = 0; kk < 2; ++kk) {
        v16h bf = load_frag(krow, 32 * kk, lane);
        sacc[0][nh] = wmma_f16(qa[0][kk], bf, sacc[0][nh]);
        sacc[1][nh] = wmma_f16(qa[1][kk], bf, sacc[1][nh]);
      }
    }
    // ---- P = mask * exp(scale*(q2+k2-2s)), stage to LDS in [m][key] ----
#pragma unroll
    for (int nh = 0; nh < 2; ++nh) {
      int keyl = 16 * nh + ln;            // D layout: n = lane&15
      int kg   = kb + keyl;
      float k2v = k2row[kg];
#pragma unroll
      for (int mt = 0; mt < 2; ++mt)
#pragma unroll
        for (int r = 0; r < 8; ++r) {
          int mrow = 16 * mt + r + 8 * h2;
          float d2 = q2v[mt][r] + k2v - 2.0f * sacc[mt][nh][r];
          float p  = (kg <= q0 + mrow) ? __expf(SCALE * d2) : 0.0f;
          myp[mrow * 32 + keyl] = (_Float16)p;
        }
    }
    // Intra-wave LDS RAW: drain DScnt before reloading P in A-layout.
    asm volatile("s_wait_dscnt 0" ::: "memory");
    v16h pa0 = load_frag(myp + ln * 32, 0, lane);
    v16h pa1 = load_frag(myp + (16 + ln) * 32, 0, lane);
    // ---- Y += P @ V ----
#pragma unroll
    for (int j = 0; j < 4; ++j) {
      yacc[0][j] = wmma_f16(pa0, vf[j], yacc[0][j]);
      yacc[1][j] = wmma_f16(pa1, vf[j], yacc[1][j]);
    }
  }
  // write Y back to (B,T,C) f16, head-interleaved columns
#pragma unroll
  for (int mt = 0; mt < 2; ++mt)
#pragma unroll
    for (int j = 0; j < 4; ++j) {
      int col = head * HD + 16 * j + ln;
#pragma unroll
      for (int r = 0; r < 8; ++r) {
        int t = q0 + 16 * mt + r + 8 * h2;
        yh[((size_t)b_ * T + t) * C + col] = (_Float16)yacc[mt][j][r];
      }
    }
}

// ---------------- output projection: out = yh @ WtP^T + b_proj (f32) -------

__global__ __launch_bounds__(256)
void proj_gemm_kernel(const _Float16* __restrict__ yh,
                      const _Float16* __restrict__ WtP,  // [C][C] transposed
                      const float* __restrict__ bias, float* __restrict__ out) {
  const int lane = threadIdx.x & 31;
  const int wave = threadIdx.x >> 5;
  const int ln   = lane & 15;
  const int m0   = blockIdx.x * 256 + wave * 32;
  const int nb   = blockIdx.y * 64;
  v8f acc[8] = {v8f{}, v8f{}, v8f{}, v8f{}, v8f{}, v8f{}, v8f{}, v8f{}};
  gemm_loop_32x64(yh + (size_t)(m0 + ln) * C,
                  yh + (size_t)(m0 + 16 + ln) * C,
                  WtP + (size_t)(nb + ln) * C,
                  WtP + (size_t)(nb + 16 + ln) * C,
                  WtP + (size_t)(nb + 32 + ln) * C,
                  WtP + (size_t)(nb + 48 + ln) * C, acc, lane);
#pragma unroll
  for (int mt = 0; mt < 2; ++mt)
#pragma unroll
    for (int j = 0; j < 4; ++j) {
      int col = nb + 16 * j + ln;
      float bv = bias[col];
#pragma unroll
      for (int r = 0; r < 8; ++r) {
        int m = m0 + 16 * mt + r + 8 * (lane >> 4);
        out[(size_t)m * C + col] = acc[mt * 4 + j][r] + bv;
      }
    }
}

// ---------------- launch ----------------

extern "C" void kernel_launch(void* const* d_in, const int* in_sizes, int n_in,
                              void* d_out, int out_size, void* d_ws, size_t ws_size,
                              hipStream_t stream) {
  (void)in_sizes; (void)n_in; (void)out_size; (void)ws_size;
  const float* x  = (const float*)d_in[0];
  const float* Wa = (const float*)d_in[1];
  const float* ba = (const float*)d_in[2];
  const float* Wp = (const float*)d_in[3];
  const float* bp = (const float*)d_in[4];

  char* ws = (char*)d_ws;
  size_t off = 0;
  auto alloc = [&](size_t bytes) -> char* {
    char* p = ws + off;
    off += (bytes + 255) & ~(size_t)255;
    return p;
  };
  _Float16* xh  = (_Float16*)alloc((size_t)M * C * 2);       // 16 MB
  _Float16* WtA = (_Float16*)alloc((size_t)N_QKV * C * 2);   //  6 MB
  _Float16* WtP = (_Float16*)alloc((size_t)C * C * 2);       //  2 MB
  _Float16* qh  = (_Float16*)alloc((size_t)M * C * 2);       // 16 MB
  _Float16* kh  = (_Float16*)alloc((size_t)M * C * 2);       // 16 MB
  _Float16* vth = (_Float16*)alloc((size_t)M * C * 2);       // 16 MB
  _Float16* yh  = (_Float16*)alloc((size_t)M * C * 2);       // 16 MB
  float* q2 = (float*)alloc((size_t)Bb * H * T * 4);         // 0.5 MB
  float* k2 = (float*)alloc((size_t)Bb * H * T * 4);         // 0.5 MB
  // total ~93 MB of workspace

  cvt_f32_f16<<<4096, 256, 0, stream>>>(x, xh, M * C);
  transpose_f32_f16<<<dim3(N_QKV / 32, C / 32), dim3(32, 32), 0, stream>>>(
      Wa, WtA, C, N_QKV);
  transpose_f32_f16<<<dim3(C / 32, C / 32), dim3(32, 32), 0, stream>>>(
      Wp, WtP, C, C);
  qkv_gemm_kernel<<<dim3(M / 256, N_QKV / 64), 256, 0, stream>>>(
      xh, WtA, ba, qh, kh, vth);
  rownorm_kernel<<<(Bb * H * T) / 256, 256, 0, stream>>>(qh, q2, Bb * H * T);
  rownorm_kernel<<<(Bb * H * T) / 256, 256, 0, stream>>>(kh, k2, Bb * H * T);
  attn_kernel<<<dim3(T / 256, Bb * H), 256, 0, stream>>>(qh, kh, vth, q2, k2, yh);
  proj_gemm_kernel<<<dim3(M / 256, C / 64), 256, 0, stream>>>(
      yh, WtP, bp, (float*)d_out);
}
